// GCN_75969381531812
// MI455X (gfx1250) — matrix operand order repaired and checked by reading, not compiled, not run on hardware
//
#include <hip/hip_runtime.h>
#include <hip/hip_bf16.h>

typedef __attribute__((ext_vector_type(16))) _Float16 v16h;
typedef __attribute__((ext_vector_type(8)))  float    v8f;

#define N_NODES 100000
#define N_EDGES 1600000
#define N_GRAPHS 64
#define HID 64

// ---------------- degree / dinv ----------------
__global__ void k_deg_init(float* d, int n) {
    int i = blockIdx.x * blockDim.x + threadIdx.x;
    if (i < n) d[i] = 1.0f;   // self-loop
}
__global__ void k_deg_accum(const int* __restrict__ col, float* d, int e) {
    int i = blockIdx.x * blockDim.x + threadIdx.x;
    if (i < e) atomicAdd(&d[col[i]], 1.0f);
}
__global__ void k_deg_finish(float* d, int n) {
    int i = blockIdx.x * blockDim.x + threadIdx.x;
    if (i < n) d[i] = rsqrtf(d[i]);
}

// ---------------- weight -> WMMA B-fragment order (f16) ----------------
// b-fragment element j of lane L for (kstep ks, ntile nt):
//   K = ks*32 + (L>>4)*16 + j , N = nt*16 + (L&15)
// stored contiguously: wf[(((ks*4)+nt)*32 + L)*16 + j]
__global__ void k_wfrag(const float* __restrict__ W, _Float16* __restrict__ wf, int K) {
    int total = (K / 32) * 4 * 32 * 16;
    int t = blockIdx.x * blockDim.x + threadIdx.x;
    if (t >= total) return;
    int j    = t & 15;
    int lane = (t >> 4) & 31;
    int nt   = (t >> 9) & 3;
    int ks   = t >> 11;
    int k = ks * 32 + (lane >> 4) * 16 + j;
    int n = nt * 16 + (lane & 15);
    wf[t] = (_Float16)W[(size_t)k * HID + n];
}

// ---------------- GEMM: H[n,64] = X[n,K] @ W[K,64] via v_wmma_f32_16x16x32_f16 ----
__global__ __launch_bounds__(256)
void k_gemm_wmma(const float* __restrict__ X, const _Float16* __restrict__ wf,
                 float* __restrict__ H, int K, int ntiles) {
    int wave = threadIdx.x >> 5;
    int lane = threadIdx.x & 31;
    int tile = blockIdx.x * (blockDim.x >> 5) + wave;
    if (tile >= ntiles) return;            // wave-uniform exit: EXEC stays all-ones
    int m    = lane & 15;
    int half = lane >> 4;

    const float* xrow = X + (size_t)(tile * 16 + m) * K;

    v8f acc0 = {}, acc1 = {}, acc2 = {}, acc3 = {};
    int nks = K >> 5;
    for (int ks = 0; ks < nks; ++ks) {
        int kb = ks * 32;
        // A fragment: contiguous k order, K = kb + half*16 + j (matches wf packing)
        const float* ap = xrow + kb + half * 16;
        float4 a0 = *(const float4*)(ap);
        float4 a1 = *(const float4*)(ap + 4);
        float4 a2 = *(const float4*)(ap + 8);
        float4 a3 = *(const float4*)(ap + 12);
        v16h a;
        a[0]  = (_Float16)a0.x; a[1]  = (_Float16)a0.y; a[2]  = (_Float16)a0.z; a[3]  = (_Float16)a0.w;
        a[4]  = (_Float16)a1.x; a[5]  = (_Float16)a1.y; a[6]  = (_Float16)a1.z; a[7]  = (_Float16)a1.w;
        a[8]  = (_Float16)a2.x; a[9]  = (_Float16)a2.y; a[10] = (_Float16)a2.z; a[11] = (_Float16)a2.w;
        a[12] = (_Float16)a3.x; a[13] = (_Float16)a3.y; a[14] = (_Float16)a3.z; a[15] = (_Float16)a3.w;

        const _Float16* wfk = wf + ((size_t)ks * 4 * 32) * 16;
        v16h b0 = *(const v16h*)(wfk + ((size_t)(0 * 32) + lane) * 16);
        v16h b1 = *(const v16h*)(wfk + ((size_t)(1 * 32) + lane) * 16);
        v16h b2 = *(const v16h*)(wfk + ((size_t)(2 * 32) + lane) * 16);
        v16h b3 = *(const v16h*)(wfk + ((size_t)(3 * 32) + lane) * 16);

        acc0 = __builtin_amdgcn_wmma_f32_16x16x32_f16(false, a, false, b0, (short)0, acc0, false, false);
        acc1 = __builtin_amdgcn_wmma_f32_16x16x32_f16(false, a, false, b1, (short)0, acc1, false, false);
        acc2 = __builtin_amdgcn_wmma_f32_16x16x32_f16(false, a, false, b2, (short)0, acc2, false, false);
        acc3 = __builtin_amdgcn_wmma_f32_16x16x32_f16(false, a, false, b3, (short)0, acc3, false, false);
    }

    // C/D layout: element v -> row M = half*8 + v, col N = lane&15 (+16*nt)
    float* out = H + (size_t)(tile * 16) * HID;
    int n0 = lane & 15;
    #pragma unroll
    for (int v = 0; v < 8; ++v) {
        size_t r = (size_t)(half * 8 + v) * HID;
        out[r +  0 + n0] = acc0[v];
        out[r + 16 + n0] = acc1[v];
        out[r + 32 + n0] = acc2[v];
        out[r + 48 + n0] = acc3[v];
    }
}

// ---------------- zero fill ----------------
__global__ void k_zero(float* p, int n) {
    int i = blockIdx.x * blockDim.x + threadIdx.x;
    if (i < n) p[i] = 0.0f;
}

// ---------------- edge scatter: agg[col] += h[row] * dinv[row]*dinv[col] --------
// 16 threads per edge, float4 each -> 256B contiguous per edge, L2-resident data
__global__ void k_scatter(const int* __restrict__ row, const int* __restrict__ col,
                          const float* __restrict__ dinv, const float* __restrict__ h,
                          float* __restrict__ agg, int E) {
    int t = blockIdx.x * blockDim.x + threadIdx.x;
    int e = t >> 4;
    if (e >= E) return;
    int c4 = (t & 15) * 4;
    int r = row[e];
    int c = col[e];
    float coef = dinv[r] * dinv[c];
    float4 hv = *(const float4*)(h + (size_t)r * HID + c4);
    float* ap = agg + (size_t)c * HID + c4;
    atomicAdd(ap + 0, hv.x * coef);
    atomicAdd(ap + 1, hv.y * coef);
    atomicAdd(ap + 2, hv.z * coef);
    atomicAdd(ap + 3, hv.w * coef);
}

// ---------------- finalize (in place into agg -> becomes next layer input) -----
__global__ void k_finalize(float* __restrict__ agg, const float* __restrict__ h,
                           const float* __restrict__ dinv, const float* __restrict__ b,
                           int n, int relu) {
    int t = blockIdx.x * blockDim.x + threadIdx.x;
    if (t >= n * HID) return;
    int i = t >> 6, f = t & 63;
    float d = dinv[i];
    float v = agg[t] + h[t] * d * d + b[f];
    if (relu) v = fmaxf(v, 0.0f);
    agg[t] = v;
}

// ---------------- mean pool ----------------
__global__ void k_pool(const float* __restrict__ feat, const int* __restrict__ batch,
                       float* __restrict__ pooled, float* __restrict__ counts, int n) {
    int t = blockIdx.x * blockDim.x + threadIdx.x;
    if (t >= n * HID) return;
    int i = t >> 6, f = t & 63;
    int g = batch[i];
    atomicAdd(&pooled[g * HID + f], feat[t]);
    if (f == 0) atomicAdd(&counts[g], 1.0f);
}

// ---------------- head: out[g,o] = (pooled[g]/cnt[g]) @ Wlin + blin ------------
__global__ void k_head(const float* __restrict__ pooled, const float* __restrict__ counts,
                       const float* __restrict__ Wlin, const float* __restrict__ blin,
                       float* __restrict__ out) {
    int t = threadIdx.x;
    if (t >= N_GRAPHS * 2) return;
    int g = t >> 1, o = t & 1;
    float inv = 1.0f / fmaxf(counts[g], 1.0f);
    float s = 0.0f;
    #pragma unroll
    for (int f = 0; f < HID; ++f)
        s += pooled[g * HID + f] * Wlin[f * 2 + o];
    out[t] = s * inv + blin[o];
}

extern "C" void kernel_launch(void* const* d_in, const int* in_sizes, int n_in,
                              void* d_out, int out_size, void* d_ws, size_t ws_size,
                              hipStream_t stream) {
    const float* x     = (const float*)d_in[0];
    const int*   ei    = (const int*)d_in[1];
    const int*   row   = ei;
    const int*   col   = ei + N_EDGES;
    const int*   batch = (const int*)d_in[2];
    const float* W1    = (const float*)d_in[3];
    const float* b1    = (const float*)d_in[4];
    const float* W2    = (const float*)d_in[5];
    const float* b2    = (const float*)d_in[6];
    const float* W3    = (const float*)d_in[7];
    const float* b3    = (const float*)d_in[8];
    const float* Wlin  = (const float*)d_in[9];
    const float* blin  = (const float*)d_in[10];
    float* out = (float*)d_out;

    // workspace carve (~52 MB), 128B-aligned slices
    float* dinv = (float*)d_ws;                           // 100032 floats (padded)
    float* h    = dinv + 100032;                          // N*64
    float* agg  = h + (size_t)N_NODES * HID;              // N*64
    _Float16* wf1 = (_Float16*)(agg + (size_t)N_NODES * HID); // 8192 halves (K=128)
    _Float16* wf2 = wf1 + 8192;                           // 4096 halves (K=64)
    _Float16* wf3 = wf2 + 4096;                           // 4096 halves
    float* pooled = (float*)(wf3 + 4096);                 // 64*64
    float* counts = pooled + N_GRAPHS * HID;              // 64

    const int B = 256;
    int nF = N_NODES * HID;       // 6,400,000
    int ntiles = N_NODES / 16;    // 6250 (exact)
    int gemm_blocks = (ntiles + 7) / 8;

    // degrees -> dinv
    k_deg_init  <<<(N_NODES + B - 1) / B, B, 0, stream>>>(dinv, N_NODES);
    k_deg_accum <<<(N_EDGES + B - 1) / B, B, 0, stream>>>(col, dinv, N_EDGES);
    k_deg_finish<<<(N_NODES + B - 1) / B, B, 0, stream>>>(dinv, N_NODES);

    // weights -> WMMA fragment order
    k_wfrag<<<(8192 + B - 1) / B, B, 0, stream>>>(W1, wf1, 128);
    k_wfrag<<<(4096 + B - 1) / B, B, 0, stream>>>(W2, wf2, 64);
    k_wfrag<<<(4096 + B - 1) / B, B, 0, stream>>>(W3, wf3, 64);

    // ---- layer 1 (input x, K=128, relu) ----
    k_gemm_wmma<<<gemm_blocks, B, 0, stream>>>(x, wf1, h, 128, ntiles);
    k_zero     <<<(nF + B - 1) / B, B, 0, stream>>>(agg, nF);
    k_scatter  <<<((N_EDGES * 16) + B - 1) / B, B, 0, stream>>>(row, col, dinv, h, agg, N_EDGES);
    k_finalize <<<(nF + B - 1) / B, B, 0, stream>>>(agg, h, dinv, b1, N_NODES, 1);

    // ---- layer 2 (input agg, K=64, relu) ----
    k_gemm_wmma<<<gemm_blocks, B, 0, stream>>>(agg, wf2, h, 64, ntiles);
    k_zero     <<<(nF + B - 1) / B, B, 0, stream>>>(agg, nF);
    k_scatter  <<<((N_EDGES * 16) + B - 1) / B, B, 0, stream>>>(row, col, dinv, h, agg, N_EDGES);
    k_finalize <<<(nF + B - 1) / B, B, 0, stream>>>(agg, h, dinv, b2, N_NODES, 1);

    // ---- layer 3 (input agg, K=64, no relu) ----
    k_gemm_wmma<<<gemm_blocks, B, 0, stream>>>(agg, wf3, h, 64, ntiles);
    k_zero     <<<(nF + B - 1) / B, B, 0, stream>>>(agg, nF);
    k_scatter  <<<((N_EDGES * 16) + B - 1) / B, B, 0, stream>>>(row, col, dinv, h, agg, N_EDGES);
    k_finalize <<<(nF + B - 1) / B, B, 0, stream>>>(agg, h, dinv, b3, N_NODES, 0);

    // ---- mean pool + head ----
    k_zero<<<(N_GRAPHS * HID + N_GRAPHS + B - 1) / B, B, 0, stream>>>(pooled, N_GRAPHS * HID + N_GRAPHS);
    k_pool<<<(nF + B - 1) / B, B, 0, stream>>>(agg, batch, pooled, counts, N_NODES);
    k_head<<<1, 128, 0, stream>>>(pooled, counts, Wlin, blin, out);
}
